// MultiHeadSelfAttention_6339371729709
// MI455X (gfx1250) — matrix operand order repaired and checked by reading, not compiled, not run on hardware
//
#include <hip/hip_runtime.h>
#include <hip/hip_bf16.h>

typedef __bf16 bf16_t;
typedef __attribute__((ext_vector_type(16))) __bf16 v16bf;
typedef __attribute__((ext_vector_type(8)))  __bf16 v8bf;
typedef __attribute__((ext_vector_type(8)))  float  v8f;

#define EMBED 512
#define SEQ   4096
#define HEADS 8
#define HD    64

// ---------------- WMMA helpers (CDNA5 16x16x32 bf16, f32 accum) -------------

static __device__ inline v8f wmma_bf16(v16bf a, v16bf b, v8f c) {
  // (neg_a, A, neg_b, B, c_mod, C, reuse_a, reuse_b)
  return __builtin_amdgcn_wmma_f32_16x16x32_bf16(false, a, false, b, (short)0, c,
                                                 false, false);
}

// A fragment: 16x32 bf16 tile at (m0,k0) of row-major [*, ld] matrix.
// lane: M = lane&15, half = lane>>4; element e<8 -> K = half*8+e,
// e>=8 -> K = 16+half*8+(e-8). Two contiguous 16B loads per lane.
static __device__ inline v16bf load_a_frag(const bf16_t* p, int ld, int m0, int k0) {
  int lane = threadIdx.x & 31;
  int hf = lane >> 4;
  const bf16_t* row = p + (size_t)(m0 + (lane & 15)) * ld + k0 + hf * 8;
  v8bf lo = *(const v8bf*)row;
  v8bf hi = *(const v8bf*)(row + 16);
  v16bf a;
#pragma unroll
  for (int e = 0; e < 8; ++e) { a[e] = lo[e]; a[e + 8] = hi[e]; }
  return a;
}

// B fragment: 32x16 (KxN) tile; source stored N-major (row-major [N, K], ld=K
// stride). lane: N = lane&15, half = lane>>4; element e<8 -> K = half*8+e,
// e>=8 -> K = 16+half*8+(e-8). Two contiguous 16B loads per lane.
static __device__ inline v16bf load_b_frag(const bf16_t* p, int ld, int n0, int k0) {
  int lane = threadIdx.x & 31;
  const bf16_t* row = p + (size_t)(n0 + (lane & 15)) * ld + k0 + (lane >> 4) * 8;
  v8bf lo = *(const v8bf*)row;
  v8bf hi = *(const v8bf*)(row + 16);
  v16bf b;
#pragma unroll
  for (int e = 0; e < 8; ++e) { b[e] = lo[e]; b[e + 8] = hi[e]; }
  return b;
}

// ---------------- fp32 -> bf16 conversion -----------------------------------

__global__ void cvt_kernel(const float* __restrict__ in, bf16_t* __restrict__ out, int n) {
  int i = blockIdx.x * blockDim.x + threadIdx.x;
  if (i < n) out[i] = (bf16_t)in[i];
}

// ---------------- QKV projection: C = X @ W^T -------------------------------
// X bf16 [8192,512]; W bf16 [512,512] row-major (== N-major for W^T GEMM).
// Q,K written as [b,h,s,64]; V written transposed as [b,h,64,s].
// block = 128 threads = 4 waves; wave computes 64x64; grid (64,4,3).

__global__ __launch_bounds__(128) void qkv_kernel(
    const bf16_t* __restrict__ X,
    const bf16_t* __restrict__ Wq, const bf16_t* __restrict__ Wk,
    const bf16_t* __restrict__ Wv,
    bf16_t* __restrict__ Q, bf16_t* __restrict__ K, bf16_t* __restrict__ Vt) {
  int wave = threadIdx.x >> 5;
  int m0 = blockIdx.x * 128 + (wave >> 1) * 64;
  int n0 = blockIdx.y * 128 + (wave & 1) * 64;
  const bf16_t* W = (blockIdx.z == 0) ? Wq : (blockIdx.z == 1) ? Wk : Wv;

  v8f acc[4][4];
#pragma unroll
  for (int i = 0; i < 4; ++i)
#pragma unroll
    for (int j = 0; j < 4; ++j)
#pragma unroll
      for (int e = 0; e < 8; ++e) acc[i][j][e] = 0.f;

  for (int k0 = 0; k0 < EMBED; k0 += 32) {
    v16bf a[4], b[4];
#pragma unroll
    for (int i = 0; i < 4; ++i) a[i] = load_a_frag(X, EMBED, m0 + 16 * i, k0);
#pragma unroll
    for (int j = 0; j < 4; ++j) b[j] = load_b_frag(W, EMBED, n0 + 16 * j, k0);
#pragma unroll
    for (int i = 0; i < 4; ++i)
#pragma unroll
      for (int j = 0; j < 4; ++j)
        acc[i][j] = wmma_bf16(a[i], b[j], acc[i][j]);
  }

  int lane = threadIdx.x & 31;
  int ln = lane & 15, hf = lane >> 4;
  if (blockIdx.z < 2) {
    bf16_t* out = (blockIdx.z == 0) ? Q : K;
#pragma unroll
    for (int i = 0; i < 4; ++i)
#pragma unroll
      for (int j = 0; j < 4; ++j) {
        int n = n0 + 16 * j + ln;
        int h = n >> 6, d = n & 63;
        int mb = m0 + 16 * i + hf * 8;  // D tile: lanes16-31 hold rows M=8..15
#pragma unroll
        for (int r = 0; r < 8; ++r) {
          int mm = mb + r;
          int bb = mm >> 12, s = mm & (SEQ - 1);
          out[((((size_t)bb * HEADS + h) * SEQ + s) << 6) + d] = (bf16_t)acc[i][j][r];
        }
      }
  } else {
    // V transposed: [b,h,64,4096]; rows (s) are contiguous per lane -> 16B store
#pragma unroll
    for (int i = 0; i < 4; ++i)
#pragma unroll
      for (int j = 0; j < 4; ++j) {
        int n = n0 + 16 * j + ln;
        int h = n >> 6, d = n & 63;
        int mb = m0 + 16 * i + hf * 8;
        int bb = mb >> 12, s = mb & (SEQ - 1);
        v8bf pk;
#pragma unroll
        for (int r = 0; r < 8; ++r) pk[r] = (bf16_t)acc[i][j][r];
        *(v8bf*)(Vt + (((size_t)bb * HEADS + h) * HD + d) * SEQ + s) = pk;
      }
  }
}

// ---------------- Flash attention -------------------------------------------
// wave handles 16 queries of one (b,h); online softmax over 4096 keys in
// 32-key steps; O accum 16x64 f32 (4 WMMA tiles). 1024 blocks x 128 threads.

__global__ __launch_bounds__(128) void attn_kernel(
    const bf16_t* __restrict__ Q, const bf16_t* __restrict__ K,
    const bf16_t* __restrict__ Vt, bf16_t* __restrict__ Att) {
  __shared__ __align__(16) bf16_t pbuf[4][16 * 32];  // per-wave P tile
  int wave = threadIdx.x >> 5;
  int lane = threadIdx.x & 31;
  int ln = lane & 15, hf = lane >> 4;
  int gw = blockIdx.x * 4 + wave;
  int b = gw >> 11;             // 2048 waves per batch
  int h = (gw >> 8) & 7;
  int s0 = (gw & 255) << 4;

  const bf16_t* qp = Q + ((size_t)b * HEADS + h) * SEQ * HD;
  const bf16_t* kp = K + ((size_t)b * HEADS + h) * SEQ * HD;
  const bf16_t* vp = Vt + ((size_t)b * HEADS + h) * HD * SEQ;

  v16bf aq0 = load_a_frag(qp, HD, s0, 0);    // Q block 16x64, held for whole loop
  v16bf aq1 = load_a_frag(qp, HD, s0, 32);

  v8f o[4];
  float m[8], l[8];
#pragma unroll
  for (int i = 0; i < 4; ++i)
#pragma unroll
    for (int e = 0; e < 8; ++e) o[i][e] = 0.f;
#pragma unroll
  for (int j = 0; j < 8; ++j) { m[j] = -1e30f; l[j] = 0.f; }

  bf16_t* pb = pbuf[wave];

  for (int kb = 0; kb < SEQ; kb += 32) {
    // scores: 16 queries x 32 keys, contract over Hd=64 (2 WMMA per key tile)
    v8f sc[2];
#pragma unroll
    for (int t = 0; t < 2; ++t) {
      v8f z;
#pragma unroll
      for (int e = 0; e < 8; ++e) z[e] = 0.f;
      z = wmma_bf16(aq0, load_b_frag(kp, HD, kb + 16 * t, 0), z);
      z = wmma_bf16(aq1, load_b_frag(kp, HD, kb + 16 * t, 32), z);
      sc[t] = z;
    }
    // row max (reduce over N across the 16-lane half; rows r = j + 8*hf)
    float alpha[8];
#pragma unroll
    for (int j = 0; j < 8; ++j) {
      float a0 = sc[0][j] * 0.125f, a1 = sc[1][j] * 0.125f;  // 1/sqrt(64)
      sc[0][j] = a0; sc[1][j] = a1;
      float v = fmaxf(a0, a1);
      v = fmaxf(v, __shfl_xor(v, 1, 32));
      v = fmaxf(v, __shfl_xor(v, 2, 32));
      v = fmaxf(v, __shfl_xor(v, 4, 32));
      v = fmaxf(v, __shfl_xor(v, 8, 32));
      float mn = fmaxf(m[j], v);
      alpha[j] = __expf(m[j] - mn);
      m[j] = mn;
    }
    // exponentials + row sums
#pragma unroll
    for (int j = 0; j < 8; ++j) {
      float p0 = __expf(sc[0][j] - m[j]);
      float p1 = __expf(sc[1][j] - m[j]);
      sc[0][j] = p0; sc[1][j] = p1;
      float v = p0 + p1;
      v += __shfl_xor(v, 1, 32);
      v += __shfl_xor(v, 2, 32);
      v += __shfl_xor(v, 4, 32);
      v += __shfl_xor(v, 8, 32);
      l[j] = l[j] * alpha[j] + v;
    }
    // rescale running O (row mapping identical to score tiles)
#pragma unroll
    for (int i = 0; i < 4; ++i)
#pragma unroll
      for (int j = 0; j < 8; ++j) o[i][j] *= alpha[j];
    // P (D layout) -> LDS row-major 16x32 -> reload as A fragment
#pragma unroll
    for (int t = 0; t < 2; ++t)
#pragma unroll
      for (int j = 0; j < 8; ++j)
        pb[(j + 8 * hf) * 32 + 16 * t + ln] = (bf16_t)sc[t][j];
    asm volatile("s_wait_dscnt 0" ::: "memory");  // in-wave LDS RAW ordering
    v8bf plo = *(const v8bf*)(pb + ln * 32 + hf * 8);
    v8bf phi = *(const v8bf*)(pb + ln * 32 + 16 + hf * 8);
    v16bf pa;
#pragma unroll
    for (int e = 0; e < 8; ++e) { pa[e] = plo[e]; pa[e + 8] = phi[e]; }
    // O += P(16x32) x V(32x64); Vt rows are N-major over hd
#pragma unroll
    for (int i = 0; i < 4; ++i)
      o[i] = wmma_bf16(pa, load_b_frag(vp, SEQ, 16 * i, kb), o[i]);
  }

  float inv[8];
#pragma unroll
  for (int j = 0; j < 8; ++j) inv[j] = 1.f / l[j];
#pragma unroll
  for (int i = 0; i < 4; ++i) {
    int d = h * 64 + 16 * i + ln;
#pragma unroll
    for (int j = 0; j < 8; ++j) {
      int sr = s0 + j + 8 * hf;
      Att[((size_t)b * SEQ + sr) * EMBED + d] = (bf16_t)(o[i][j] * inv[j]);
    }
  }
}

// ---------------- Output projection: out = Att @ Wc^T + bc (f32 out) --------

__global__ __launch_bounds__(128) void proj_kernel(
    const bf16_t* __restrict__ A, const bf16_t* __restrict__ Wc,
    const float* __restrict__ bias, float* __restrict__ out) {
  int wave = threadIdx.x >> 5;
  int m0 = blockIdx.x * 128 + (wave >> 1) * 64;
  int n0 = blockIdx.y * 128 + (wave & 1) * 64;

  v8f acc[4][4];
#pragma unroll
  for (int i = 0; i < 4; ++i)
#pragma unroll
    for (int j = 0; j < 4; ++j)
#pragma unroll
      for (int e = 0; e < 8; ++e) acc[i][j][e] = 0.f;

  for (int k0 = 0; k0 < EMBED; k0 += 32) {
    v16bf a[4], b[4];
#pragma unroll
    for (int i = 0; i < 4; ++i) a[i] = load_a_frag(A, EMBED, m0 + 16 * i, k0);
#pragma unroll
    for (int j = 0; j < 4; ++j) b[j] = load_b_frag(Wc, EMBED, n0 + 16 * j, k0);
#pragma unroll
    for (int i = 0; i < 4; ++i)
#pragma unroll
      for (int j = 0; j < 4; ++j)
        acc[i][j] = wmma_bf16(a[i], b[j], acc[i][j]);
  }

  int lane = threadIdx.x & 31;
  int ln = lane & 15, hf = lane >> 4;
#pragma unroll
  for (int i = 0; i < 4; ++i)
#pragma unroll
    for (int j = 0; j < 4; ++j) {
      int n = n0 + 16 * j + ln;
      float bn = bias[n];
      int mb = m0 + 16 * i + hf * 8;
#pragma unroll
      for (int r = 0; r < 8; ++r)
        out[(size_t)(mb + r) * EMBED + n] = acc[i][j][r] + bn;
    }
}

// ---------------- launch -----------------------------------------------------

extern "C" void kernel_launch(void* const* d_in, const int* in_sizes, int n_in,
                              void* d_out, int out_size, void* d_ws, size_t ws_size,
                              hipStream_t stream) {
  (void)in_sizes; (void)n_in; (void)out_size; (void)ws_size;
  const float* X  = (const float*)d_in[0];
  const float* Wq = (const float*)d_in[1];
  const float* Wk = (const float*)d_in[2];
  const float* Wv = (const float*)d_in[3];
  const float* Wc = (const float*)d_in[4];
  const float* bc = (const float*)d_in[5];

  char* ws = (char*)d_ws;
  const size_t XN = (size_t)2 * SEQ * EMBED;   // 4,194,304 elems
  const size_t WN = (size_t)EMBED * EMBED;     //   262,144 elems
  bf16_t* Xb   = (bf16_t*)(ws + 0);            //  8 MB
  bf16_t* Wqb  = (bf16_t*)(ws + 8388608);
  bf16_t* Wkb  = (bf16_t*)(ws + 8912896);
  bf16_t* Wvb  = (bf16_t*)(ws + 9437184);
  bf16_t* Wcb  = (bf16_t*)(ws + 9961472);
  bf16_t* Qb   = (bf16_t*)(ws + 10485760);     //  8 MB [b,h,s,64]
  bf16_t* Kb   = (bf16_t*)(ws + 18874368);     //  8 MB [b,h,s,64]
  bf16_t* Vtb  = (bf16_t*)(ws + 27262976);     //  8 MB [b,h,64,s]
  bf16_t* Attb = (bf16_t*)(ws + 35651584);     //  8 MB [b,s,512]

  cvt_kernel<<<dim3((unsigned)((XN + 255) / 256)), 256, 0, stream>>>(X, Xb, (int)XN);
  cvt_kernel<<<dim3((unsigned)((WN + 255) / 256)), 256, 0, stream>>>(Wq, Wqb, (int)WN);
  cvt_kernel<<<dim3((unsigned)((WN + 255) / 256)), 256, 0, stream>>>(Wk, Wkb, (int)WN);
  cvt_kernel<<<dim3((unsigned)((WN + 255) / 256)), 256, 0, stream>>>(Wv, Wvb, (int)WN);
  cvt_kernel<<<dim3((unsigned)((WN + 255) / 256)), 256, 0, stream>>>(Wc, Wcb, (int)WN);

  qkv_kernel<<<dim3(64, 4, 3), 128, 0, stream>>>(Xb, Wqb, Wkb, Wvb, Qb, Kb, Vtb);
  attn_kernel<<<dim3(1024), 128, 0, stream>>>(Qb, Kb, Vtb, Attb);
  proj_kernel<<<dim3(64, 4), 128, 0, stream>>>(Attb, Wcb, bc, (float*)d_out);
}
